// SingalHeadAttention_48627619725371
// MI455X (gfx1250) — matrix operand order repaired and checked by reading, not compiled, not run on hardware
//
#include <hip/hip_runtime.h>
#include <hip/hip_bf16.h>

typedef __attribute__((ext_vector_type(16))) _Float16 v16h;
typedef __attribute__((ext_vector_type(8)))  _Float16 v8h;
typedef __attribute__((ext_vector_type(8)))  float    v8f;
typedef __attribute__((ext_vector_type(4)))  int      v4i;

union V16 { v16h v; v8h h[2]; };

#define WMMA_F16(a, b, c) \
  __builtin_amdgcn_wmma_f32_16x16x32_f16(false, (a), false, (b), (short)0, (c), false, false)

constexpr int kB = 4, kS = 4096, kD = 768, kH = 64;

// -------- gfx1250 async global->LDS path (guarded; fallback = direct loads) --
#if defined(__has_builtin)
#if __has_builtin(__builtin_amdgcn_global_load_async_to_lds_b128) && \
    __has_builtin(__builtin_amdgcn_s_wait_asynccnt)
#define USE_ASYNC_LDS 1
#endif
#endif

#if defined(USE_ASYNC_LDS)
#define AS3 __attribute__((address_space(3)))
#define ASYNC_B128(gp, lp) \
  __builtin_amdgcn_global_load_async_to_lds_b128( \
      (v4i*)(void*)(gp), (AS3 v4i*)(AS3 void*)(void*)(lp), 0, 0)
#endif

// ---------------------------------------------------------------------------
// Kernel 0: convert Wq/Wk/Wv fp32 [768,64] -> f16 transposed Wt[m][col][k]
// so WMMA B-fragments become single contiguous 32-byte loads.
// ---------------------------------------------------------------------------
__global__ __launch_bounds__(256) void wconv_48627619725371(
    const float* __restrict__ Wq, const float* __restrict__ Wk,
    const float* __restrict__ Wv, _Float16* __restrict__ Wt)
{
  const int e = blockIdx.x * 256 + threadIdx.x;   // 0 .. 3*768*64-1
  const int m = e / (kD * kH);
  const int idx = e % (kD * kH);
  const int k = idx >> 6, c = idx & 63;
  const float* W = (m == 0) ? Wq : (m == 1) ? Wk : Wv;
  Wt[(size_t)m * (kD * kH) + (size_t)c * kD + k] = (_Float16)W[idx];
}

// ---------------------------------------------------------------------------
// Kernel 1: QKV projection, no LDS / no barriers.  Q,K f16 [B*S,64];
// V f16 transposed [B,64,S].  Q pre-scaled by 1/sqrt(64).
// One wave = 16 rows of X; 8 waves/block.
// ---------------------------------------------------------------------------
__global__ __launch_bounds__(256) void qkv_proj_48627619725371(
    const float* __restrict__ X,   const _Float16* __restrict__ Wt,
    const float* __restrict__ bq,  const float* __restrict__ bk,
    const float* __restrict__ bv,
    _Float16* __restrict__ Qh, _Float16* __restrict__ Kh,
    _Float16* __restrict__ Vt)
{
  const int tid   = threadIdx.x;
  const int wave  = tid >> 5;
  const int lane  = tid & 31;
  const int laneM = lane & 15;
  const int half  = lane >> 4;
  const long R0   = (long)blockIdx.x * 128 + (long)wave * 16;

  v8f acc[3][4];
  #pragma unroll
  for (int m = 0; m < 3; ++m)
    #pragma unroll
    for (int n = 0; n < 4; ++n)
      acc[m][n] = (v8f){0.f, 0.f, 0.f, 0.f, 0.f, 0.f, 0.f, 0.f};

  // Per-(m,n) W column pointers: lane reads 16 contiguous halves per k-step.
  const _Float16* wcol[3][4];
  #pragma unroll
  for (int m = 0; m < 3; ++m)
    #pragma unroll
    for (int n = 0; n < 4; ++n)
      wcol[m][n] = Wt + (size_t)m * (kD * kH) + (size_t)(n * 16 + laneM) * kD
                      + half * 16;

  const float* xbase = X + (R0 + laneM) * kD + half * 16;

  #pragma unroll 2
  for (int kb = 0; kb < kD; kb += 32) {
    // Batch-load ALL 12 B-fragments into distinct registers.
    v16h bf[3][4];
    #pragma unroll
    for (int m = 0; m < 3; ++m)
      #pragma unroll
      for (int n = 0; n < 4; ++n)
        bf[m][n] = *(const v16h*)(wcol[m][n] + kb);

    // A fragment (X tile 16x32, f32 -> f16), contiguous k = 16*half + s.
    const float* xrow = xbase + kb;
    V16 af;
    const float4 x0 = ((const float4*)xrow)[0];
    const float4 x1 = ((const float4*)xrow)[1];
    const float4 x2 = ((const float4*)(xrow + 8))[0];
    const float4 x3 = ((const float4*)(xrow + 8))[1];
    af.h[0] = (v8h){(_Float16)x0.x, (_Float16)x0.y, (_Float16)x0.z, (_Float16)x0.w,
                    (_Float16)x1.x, (_Float16)x1.y, (_Float16)x1.z, (_Float16)x1.w};
    af.h[1] = (v8h){(_Float16)x2.x, (_Float16)x2.y, (_Float16)x2.z, (_Float16)x2.w,
                    (_Float16)x3.x, (_Float16)x3.y, (_Float16)x3.z, (_Float16)x3.w};

    // Scheduling barrier: forbid sinking the fragment loads into the WMMA
    // chain (compiler otherwise serializes load->wait0->wmma twelve times).
    asm volatile("" : "+v"(bf[0][0]), "+v"(bf[0][1]), "+v"(bf[0][2]), "+v"(bf[0][3]),
                      "+v"(bf[1][0]), "+v"(bf[1][1]), "+v"(bf[1][2]), "+v"(bf[1][3]),
                      "+v"(bf[2][0]), "+v"(bf[2][1]), "+v"(bf[2][2]), "+v"(bf[2][3]));

    #pragma unroll
    for (int m = 0; m < 3; ++m)
      #pragma unroll
      for (int n = 0; n < 4; ++n)
        acc[m][n] = WMMA_F16(af.v, bf[m][n], acc[m][n]);
  }

  const int b  = (int)(R0 / kS);
  const int s0 = (int)(R0 % kS);
  #pragma unroll
  for (int n = 0; n < 4; ++n) {
    const int col = n * 16 + laneM;
    const float bqv = bq[col], bkv = bk[col], bvv = bv[col];
    #pragma unroll
    for (int r = 0; r < 8; ++r) {
      const int row = r + 8 * half;
      const long gr = R0 + row;
      Qh[gr * kH + col] = (_Float16)((acc[0][n][r] + bqv) * 0.125f);
      Kh[gr * kH + col] = (_Float16)(acc[1][n][r] + bkv);
      Vt[((long)b * kH + col) * kS + (s0 + row)] = (_Float16)(acc[2][n][r] + bvv);
    }
  }
}

// ---------------------------------------------------------------------------
// Kernel 2: streaming causal flash attention, one wave per 16-query tile,
// key tiles in pairs (32 keys).  K/V tiles double-buffered into per-wave LDS
// via GLOBAL_LOAD_ASYNC_TO_LDS_B128 when available.
// ---------------------------------------------------------------------------
__global__ __launch_bounds__(256) void flash_attn_48627619725371(
    const _Float16* __restrict__ Qh, const _Float16* __restrict__ Kh,
    const _Float16* __restrict__ Vt, float* __restrict__ Out)
{
  __shared__ __align__(64) _Float16 Plds[8][16][32];        // 8 KB
#if defined(USE_ASYNC_LDS)
  __shared__ __align__(64) _Float16 Kst[8][2][32][64];      // 64 KB
  __shared__ __align__(64) _Float16 Vst[8][2][64][32];      // 64 KB
#endif

  const int tid   = threadIdx.x;
  const int wave  = tid >> 5;
  const int lane  = tid & 31;
  const int laneM = lane & 15;
  const int half  = lane >> 4;

  const int qtg = blockIdx.x * 8 + wave;
  const int b   = qtg >> 8;
  const int qt  = qtg & 255;
  const long q0 = (long)b * kS + (long)qt * 16;

  const _Float16* Kb = Kh + (long)b * kS * kH;
  const _Float16* Vb = Vt + (long)b * kH * kS;

  // Q A-fragments (h=[0,32) and [32,64))
  const _Float16* qrow = Qh + (q0 + laneM) * kH;
  V16 qa0, qa1;
  qa0.h[0] = *(const v8h*)(qrow + half * 16);
  qa0.h[1] = *(const v8h*)(qrow + half * 16 + 8);
  qa1.h[0] = *(const v8h*)(qrow + 32 + half * 16);
  qa1.h[1] = *(const v8h*)(qrow + 32 + half * 16 + 8);

  v8f o[4];
  #pragma unroll
  for (int n = 0; n < 4; ++n)
    o[n] = (v8f){0.f, 0.f, 0.f, 0.f, 0.f, 0.f, 0.f, 0.f};
  float mrow[8], lrow[8];
  #pragma unroll
  for (int r = 0; r < 8; ++r) { mrow[r] = -1e30f; lrow[r] = 0.f; }

  const int npair = (qt >> 1) + 1;

#if defined(USE_ASYNC_LDS)
  // Issue 16 async B128 transfers for key-pair starting at k0n into buffer bf.
  auto issue_pair = [&](int k0n, int bf) {
    const _Float16* gK = Kb + (long)(k0n + lane) * kH;       // K row per lane
    _Float16* lK = &Kst[wave][bf][lane][0];
    #pragma unroll
    for (int c = 0; c < 8; ++c)                              // 32 rows x 128 B
      ASYNC_B128(gK + c * 8, lK + c * 8);
    #pragma unroll
    for (int j = 0; j < 8; ++j) {                            // 64 rows x 64 B
      const int flat = j * 32 + lane;
      const int row = flat >> 2, ch = flat & 3;
      ASYNC_B128(Vb + (long)row * kS + k0n + ch * 8,
                 &Vst[wave][bf][row][ch * 8]);
    }
  };
  issue_pair(0, 0);
  int buf = 0;
#endif

  for (int ip = 0; ip < npair; ++ip) {
    const int k0 = ip * 32;

#if defined(USE_ASYNC_LDS)
    if (ip + 1 < npair) {
      issue_pair(k0 + 32, buf ^ 1);               // prefetch next pair
      __builtin_amdgcn_s_wait_asynccnt(16);       // current pair's 16 done
    } else {
      __builtin_amdgcn_s_wait_asynccnt(0);
    }
#else
    if (ip + 1 < npair) {
      __builtin_prefetch(Kb + (long)(k0 + 32 + laneM) * kH, 0, 1);
      __builtin_prefetch(Vb + (long)laneM * kS + k0 + 32, 0, 1);
    }
#endif

    // Hoist ALL K and V fragments now: K feeds the score WMMAs immediately;
    // V loads overlap the softmax VALU/trans work below.
    v16h kf[2][2], vf[4];
    #pragma unroll
    for (int t = 0; t < 2; ++t) {
#if defined(USE_ASYNC_LDS)
      const _Float16* krow = &Kst[wave][buf][t * 16 + laneM][0];
#else
      const _Float16* krow = Kb + (long)(k0 + t * 16 + laneM) * kH;
#endif
      kf[t][0] = *(const v16h*)(krow + half * 16);
      kf[t][1] = *(const v16h*)(krow + 32 + half * 16);
    }
    #pragma unroll
    for (int n = 0; n < 4; ++n) {
#if defined(USE_ASYNC_LDS)
      vf[n] = *(const v16h*)&Vst[wave][buf][n * 16 + laneM][half * 16];
#else
      vf[n] = *(const v16h*)(Vb + (long)(n * 16 + laneM) * kS + k0 + half * 16);
#endif
    }
    // Keep K/V fragment loads hoisted (don't sink V reads below softmax).
    asm volatile("" : "+v"(kf[0][0]), "+v"(kf[0][1]), "+v"(kf[1][0]), "+v"(kf[1][1]),
                      "+v"(vf[0]), "+v"(vf[1]), "+v"(vf[2]), "+v"(vf[3]));

    // Scores: S = Q (16x64) . K^T (64x16) for two 16-key tiles
    v8f s[2];
    #pragma unroll
    for (int t = 0; t < 2; ++t) {
      v8f st = (v8f){0.f, 0.f, 0.f, 0.f, 0.f, 0.f, 0.f, 0.f};
      st = WMMA_F16(qa0.v, kf[t][0], st);
      st = WMMA_F16(qa1.v, kf[t][1], st);
      s[t] = st;
    }

    // Causal mask (wave-uniform branch, per-lane select)
    #pragma unroll
    for (int t = 0; t < 2; ++t) {
      const int kt = ip * 2 + t;
      if (kt >= qt) {
        #pragma unroll
        for (int r = 0; r < 8; ++r) {
          const int qq = qt * 16 + r + 8 * half;
          const int kk = kt * 16 + laneM;
          s[t][r] = (kk > qq) ? -1e30f : s[t][r];
        }
      }
    }

    // Online softmax; row r+8*half lives in the 16 lanes of this half.
    float rmax[8], corr[8];
    #pragma unroll
    for (int r = 0; r < 8; ++r) {
      float v = fmaxf(s[0][r], s[1][r]);
      #pragma unroll
      for (int off = 8; off >= 1; off >>= 1)
        v = fmaxf(v, __shfl_xor(v, off, 32));
      rmax[r] = v;
    }
    #pragma unroll
    for (int r = 0; r < 8; ++r) {
      const float mnew = fmaxf(mrow[r], rmax[r]);
      corr[r] = __expf(mrow[r] - mnew);
      mrow[r] = mnew;
      s[0][r] = __expf(s[0][r] - mnew);
      s[1][r] = __expf(s[1][r] - mnew);
      float v = s[0][r] + s[1][r];
      #pragma unroll
      for (int off = 8; off >= 1; off >>= 1)
        v += __shfl_xor(v, off, 32);
      lrow[r] = lrow[r] * corr[r] + v;
    }
    #pragma unroll
    for (int n = 0; n < 4; ++n)
      #pragma unroll
      for (int r = 0; r < 8; ++r)
        o[n][r] *= corr[r];

    // C-layout -> A-layout for P via per-wave LDS patch
    _Float16 (*P)[32] = Plds[wave];
    #pragma unroll
    for (int t = 0; t < 2; ++t)
      #pragma unroll
      for (int r = 0; r < 8; ++r)
        P[r + 8 * half][t * 16 + laneM] = (_Float16)s[t][r];

    V16 pa;
    pa.h[0] = *(const v8h*)&P[laneM][half * 16];
    pa.h[1] = *(const v8h*)&P[laneM][half * 16 + 8];

    // O += P (16x32) . V (32x16) for 4 h-tiles
    #pragma unroll
    for (int n = 0; n < 4; ++n)
      o[n] = WMMA_F16(pa.v, vf[n], o[n]);

#if defined(USE_ASYNC_LDS)
    buf ^= 1;
#endif
  }

  // Normalize and store fp32 output [B*S, 64]
  #pragma unroll
  for (int r = 0; r < 8; ++r) {
    const float inv = 1.0f / lrow[r];
    const long row = q0 + r + 8 * half;
    #pragma unroll
    for (int n = 0; n < 4; ++n)
      Out[row * kH + n * 16 + laneM] = o[n][r] * inv;
  }
}

// ---------------------------------------------------------------------------
extern "C" void kernel_launch(void* const* d_in, const int* in_sizes, int n_in,
                              void* d_out, int out_size, void* d_ws, size_t ws_size,
                              hipStream_t stream) {
  const float* X  = (const float*)d_in[0];
  const float* Wq = (const float*)d_in[1];
  const float* bq = (const float*)d_in[2];
  const float* Wk = (const float*)d_in[3];
  const float* bk = (const float*)d_in[4];
  const float* Wv = (const float*)d_in[5];
  const float* bv = (const float*)d_in[6];
  float* Out = (float*)d_out;

  const size_t nElem = (size_t)kB * kS * kH;      // 1,048,576
  _Float16* Qh = (_Float16*)d_ws;
  _Float16* Kh = Qh + nElem;
  _Float16* Vt = Kh + nElem;
  _Float16* Wt = Vt + nElem;                      // 3*768*64 halves (288 KB)

  wconv_48627619725371<<<dim3((3 * kD * kH) / 256), dim3(256), 0, stream>>>(
      Wq, Wk, Wv, Wt);
  qkv_proj_48627619725371<<<dim3((kB * kS) / 128), dim3(256), 0, stream>>>(
      X, Wt, bq, bk, bv, Qh, Kh, Vt);
  flash_attn_48627619725371<<<dim3((kB * (kS / 16)) / 8), dim3(256), 0, stream>>>(
      Qh, Kh, Vt, Out);
}
